// Mamba3_18313740550522
// MI455X (gfx1250) — matrix operand order repaired and checked by reading, not compiled, not run on hardware
//
#include <hip/hip_runtime.h>
#include <hip/hip_bf16.h>

#define B_SZ 2
#define L 512
#define BL (B_SZ * L)
#define D_MODEL 1024
#define D_STATE 128
#define HEADDIM 64
#define D_INNER 2048
#define D_SSM 2048
#define NHEADS 32
#define HALF 64
#define D_IN_PROJ 6464
#define EPS 1e-5f

typedef __attribute__((ext_vector_type(16))) __bf16 v16bf;
typedef __attribute__((ext_vector_type(8)))  float  v8f;

union Frag { v16bf v; unsigned u[8]; };

__device__ __forceinline__ float sigmoidf(float x) { return 1.0f / (1.0f + expf(-x)); }
__device__ __forceinline__ float siluf(float x)    { return x * sigmoidf(x); }
__device__ __forceinline__ float softplusf(float x){ return x > 20.0f ? x : log1pf(expf(x)); }

// CDNA5 async global->LDS copy (16 bytes), tracked by ASYNCcnt (ISA 08_async_tensor).
__device__ __forceinline__ void async_copy_b128(unsigned lds_off, const void* gptr) {
    unsigned long long ga = (unsigned long long)gptr;
    asm volatile("global_load_async_to_lds_b128 %0, %1, off"
                 :: "v"(lds_off), "v"(ga) : "memory");
}
__device__ __forceinline__ void wait_asynccnt0() {
    asm volatile("s_wait_asynccnt 0x0" ::: "memory");
}

// ---------------- conversion kernels ----------------
__global__ void k_cvt(const float* __restrict__ s, __bf16* __restrict__ d, size_t n) {
    size_t i = (size_t)blockIdx.x * blockDim.x + threadIdx.x;
    size_t stride = (size_t)gridDim.x * blockDim.x;
    for (; i < n; i += stride) d[i] = (__bf16)s[i];
}

// src [K,N] row-major f32 -> dst [N,K] row-major bf16 (transpose)
__global__ void k_cvtT(const float* __restrict__ s, __bf16* __restrict__ d, int K, int N) {
    size_t n = (size_t)K * N;
    size_t i = (size_t)blockIdx.x * blockDim.x + threadIdx.x;
    size_t stride = (size_t)gridDim.x * blockDim.x;
    for (; i < n; i += stride) {
        size_t col = i / K;       // output row = original column
        size_t k   = i - col * K; // output col = original row
        d[i] = (__bf16)s[k * N + col];
    }
}

// ---------------- WMMA GEMM with async double-buffered LDS staging ----------------
// C[M,N] = A[M,K] * Bt[N,K]^T.  A, Bt bf16 row-major; C f32 row-major.
// Block: 256 thr (8 waves), tile 128(M) x 64(N); wave tile 32x32 (2x2 wmma).
// Requires M%128==0, N%64==0, K%32==0 (grids launched exactly).
#define KSTEP 32
#define LDS_STRIDE 40   // 32 halfs + 8 pad -> 80B row stride, conflict-free b128 reads

__global__ void k_gemm(const __bf16* __restrict__ A, const __bf16* __restrict__ Bt,
                       float* __restrict__ C, int M, int N, int K) {
    __shared__ __align__(16) __bf16 shA[2][128 * LDS_STRIDE];
    __shared__ __align__(16) __bf16 shB[2][64 * LDS_STRIDE];

    const int t    = threadIdx.x;
    const int lane = t & 31;
    const int wave = t >> 5;
    const int wm   = wave >> 1;      // 0..3 -> 32-row band
    const int wn   = wave & 1;       // 0..1 -> 32-col band
    const int row  = lane & 15;
    const int sel  = lane >> 4;

    const int nbn = N >> 6;
    const int bm  = blockIdx.x / nbn;
    const int bn  = blockIdx.x % nbn;
    const int m0  = bm << 7;
    const int n0  = bn << 6;

    // async-copy addressing: thread t covers (row = t/4, 16B segment = t%4)
    const int crow = t >> 2;
    const int cseg = (t & 3) << 3;   // half-offset of 16B segment

    auto issue_tile = [&](int kb, int ib) {
        // A tile: 128 rows x 32 halfs
        {
            const __bf16* g0 = A + (size_t)(m0 + crow) * K + kb + cseg;
            const __bf16* g1 = A + (size_t)(m0 + crow + 64) * K + kb + cseg;
            unsigned l0 = (unsigned)(size_t)&shA[ib][crow * LDS_STRIDE + cseg];
            unsigned l1 = (unsigned)(size_t)&shA[ib][(crow + 64) * LDS_STRIDE + cseg];
            async_copy_b128(l0, g0);
            async_copy_b128(l1, g1);
        }
        // B tile: 64 rows x 32 halfs
        {
            const __bf16* g = Bt + (size_t)(n0 + crow) * K + kb + cseg;
            unsigned l = (unsigned)(size_t)&shB[ib][crow * LDS_STRIDE + cseg];
            async_copy_b128(l, g);
        }
    };

    v8f acc[2][2];
#pragma unroll
    for (int mi = 0; mi < 2; ++mi)
#pragma unroll
        for (int ni = 0; ni < 2; ++ni)
            acc[mi][ni] = (v8f){0.f, 0.f, 0.f, 0.f, 0.f, 0.f, 0.f, 0.f};

    issue_tile(0, 0);

    int ib = 0;
    for (int kb = 0; kb < K; kb += KSTEP, ib ^= 1) {
        wait_asynccnt0();
        __syncthreads();
        if (kb + KSTEP < K) issue_tile(kb + KSTEP, ib ^ 1);

        Frag a[2], b[2];
#pragma unroll
        for (int mi = 0; mi < 2; ++mi) {
            const __bf16* ar = &shA[ib][(wm * 32 + mi * 16 + row) * LDS_STRIDE];
#pragma unroll
            for (int v = 0; v < 4; ++v) {
                a[mi].u[v]     = *(const unsigned*)(ar + sel * 16 + 2 * v);       // K=sel*16 .. +7
                a[mi].u[v + 4] = *(const unsigned*)(ar + sel * 16 + 8 + 2 * v);   // K=sel*16+8 .. +15
            }
        }
#pragma unroll
        for (int ni = 0; ni < 2; ++ni) {
            const __bf16* br = &shB[ib][(wn * 32 + ni * 16 + row) * LDS_STRIDE];
#pragma unroll
            for (int v = 0; v < 8; ++v)
                b[ni].u[v] = *(const unsigned*)(br + sel * 16 + 2 * v);           // K=sel*16 .. +15
        }
#pragma unroll
        for (int mi = 0; mi < 2; ++mi)
#pragma unroll
            for (int ni = 0; ni < 2; ++ni)
                acc[mi][ni] = __builtin_amdgcn_wmma_f32_16x16x32_bf16(
                    false, a[mi].v, false, b[ni].v, (short)0, acc[mi][ni], false, false);
    }

#pragma unroll
    for (int mi = 0; mi < 2; ++mi)
#pragma unroll
        for (int ni = 0; ni < 2; ++ni)
#pragma unroll
            for (int r = 0; r < 8; ++r)
                C[(size_t)(m0 + wm * 32 + mi * 16 + r + 8 * sel) * N
                  + n0 + wn * 32 + ni * 16 + row] = acc[mi][ni][r];
}

// ---------------- per-row prep: RMS(B/C), dt/dA/lam, silu(x) ----------------
__global__ void k_prep(const float* __restrict__ zx,
                       const float* __restrict__ B_bias, const float* __restrict__ C_bias,
                       const float* __restrict__ Bnw, const float* __restrict__ Cnw,
                       const float* __restrict__ dt_bias, const float* __restrict__ A_log,
                       float* __restrict__ Bg, float* __restrict__ Cg,
                       float* __restrict__ dtb, float* __restrict__ dAb,
                       float* __restrict__ lamb, float* __restrict__ lamg,
                       float* __restrict__ sx) {
    const int r = blockIdx.x;
    const int t = threadIdx.x;
    const float* row = zx + (size_t)r * D_IN_PROJ;
    __shared__ float sB[128], sC[128], sL[32], sBC[2];

    float bv = 0.f, cv = 0.f;
    if (t < 128) {
        bv = row[4096 + t] + B_bias[t];
        cv = row[4224 + t] + C_bias[t];
        sB[t] = bv * bv;
        sC[t] = cv * cv;
    }
    if (t < 32) {
        float dtv = softplusf(row[4352 + t] + dt_bias[t]);
        float dAv = expf(-dtv * expf(A_log[t]));
        float lm  = sigmoidf(row[4384 + t]);
        dtb[r * 32 + t] = dtv;
        dAb[r * 32 + t] = dAv;
        lamb[r * 32 + t] = lm;
        sL[t] = lm;
    }
    __syncthreads();
    for (int s = 64; s > 0; s >>= 1) {
        if (t < s) { sB[t] += sB[t + s]; sC[t] += sC[t + s]; }
        __syncthreads();
    }
    if (t == 0) {
        sBC[0] = rsqrtf(sB[0] * (1.0f / 128.0f) + EPS);
        sBC[1] = rsqrtf(sC[0] * (1.0f / 128.0f) + EPS);
        float acc = 0.f;
        for (int h = 0; h < 32; ++h) acc += sL[h];
        lamg[r] = acc * (1.0f / 32.0f);
    }
    __syncthreads();
    if (t < 128) {
        Bg[(size_t)r * 128 + t] = bv * sBC[0] * Bnw[t];
        Cg[(size_t)r * 128 + t] = cv * sBC[1] * Cnw[t];
    }
    for (int i = t; i < D_SSM; i += 256)
        sx[(size_t)r * D_SSM + i] = siluf(row[D_INNER + i]);
}

// ---------------- cumsum(theta) along L: 4096 independent chains ----------------
__global__ void k_cumsum(const float* __restrict__ zx, float* __restrict__ cum) {
    int tid = blockIdx.x * blockDim.x + threadIdx.x;  // [0, 4096)
    int b = tid >> 11;
    int c = tid & 2047;
    float acc = 0.f;
    for (int l = 0; l < L; ++l) {
        size_t r = (size_t)(b * L + l);
        acc += zx[r * D_IN_PROJ + 4416 + c];
        cum[r * 2048 + c] = acc;
    }
}

// ---------------- head-mean cos/sin + RoPE on B and C ----------------
__global__ void k_rope(const float* __restrict__ cum,
                       const float* __restrict__ Bg, const float* __restrict__ Cg,
                       float* __restrict__ Br, float* __restrict__ Cr) {
    const int r = blockIdx.x;
    const int t = threadIdx.x;
    __shared__ float sc[256], ss[256];
    int j = t & 63, g = t >> 6;
    float pc = 0.f, ps = 0.f;
    for (int hh = 0; hh < 8; ++hh) {
        float ang = cum[(size_t)r * 2048 + (g * 8 + hh) * 64 + j];
        pc += cosf(ang);
        ps += sinf(ang);
    }
    sc[t] = pc; ss[t] = ps;
    __syncthreads();
    if (t < 64) {
        float c = (sc[t] + sc[t + 64] + sc[t + 128] + sc[t + 192]) * (1.0f / 32.0f);
        float s = (ss[t] + ss[t + 64] + ss[t + 128] + ss[t + 192]) * (1.0f / 32.0f);
        float b1 = Bg[(size_t)r * 128 + t], b2 = Bg[(size_t)r * 128 + 64 + t];
        Br[(size_t)r * 128 + t]      = b1 * c - b2 * s;
        Br[(size_t)r * 128 + 64 + t] = b2 * c + b1 * s;
        float c1 = Cg[(size_t)r * 128 + t], c2 = Cg[(size_t)r * 128 + 64 + t];
        Cr[(size_t)r * 128 + t]      = c1 * c - c2 * s;
        Cr[(size_t)r * 128 + 64 + t] = c2 * c + c1 * s;
    }
}

// ---------------- one-step shift mix: x_eff, B_eff ----------------
__global__ void k_shift(const float* __restrict__ sx, const float* __restrict__ Br,
                        const float* __restrict__ lam, const float* __restrict__ lamg,
                        float* __restrict__ xeff, float* __restrict__ Beff) {
    const int r = blockIdx.x;
    const int t = threadIdx.x;
    const bool first = ((r & (L - 1)) == 0);
    for (int i = t; i < D_SSM; i += 256) {
        float la = lam[r * 32 + (i >> 6)];
        float cur = sx[(size_t)r * D_SSM + i];
        float prev = first ? 0.f : sx[(size_t)(r - 1) * D_SSM + i];
        xeff[(size_t)r * D_SSM + i] = la * cur + (1.f - la) * prev;
    }
    if (t < 128) {
        float lg = lamg[r];
        float cur = Br[(size_t)r * 128 + t];
        float prev = first ? 0.f : Br[(size_t)(r - 1) * 128 + t];
        Beff[(size_t)r * 128 + t] = lg * cur + (1.f - lg) * prev;
    }
}

// ---------------- sequential SSM scan: one block per (b, head) ----------------
__global__ void k_scan(const float* __restrict__ dAb, const float* __restrict__ dtb,
                       const float* __restrict__ xeff, const float* __restrict__ Beff,
                       const float* __restrict__ Cr, const float* __restrict__ Dvec,
                       float* __restrict__ ybuf) {
    const int b = blockIdx.x >> 5;
    const int h = blockIdx.x & 31;
    const int t = threadIdx.x;
    const int p = t & 63;        // headdim index
    const int dseg = t >> 6;     // 4 segments of 32 states
    const int d0 = dseg * 32;
    __shared__ float shB[128], shC[128], shX[64], shS[2], shRed[256];
    float hreg[32];
#pragma unroll
    for (int i = 0; i < 32; ++i) hreg[i] = 0.f;
    const float Dh = Dvec[h];

    for (int l = 0; l < L; ++l) {
        const size_t r = (size_t)(b * L + l);
        if (t < 128) { shB[t] = Beff[r * 128 + t]; shC[t] = Cr[r * 128 + t]; }
        if (t < 64)  shX[t] = xeff[r * D_SSM + h * 64 + t];
        if (t == 0)  { shS[0] = dAb[r * 32 + h]; shS[1] = dtb[r * 32 + h]; }
        __syncthreads();
        const float a = shS[0];
        const float coef = shS[1] * shX[p];
        float acc = 0.f;
#pragma unroll
        for (int i = 0; i < 32; ++i) {
            float hb = fmaf(a, hreg[i], coef * shB[d0 + i]);
            hreg[i] = hb;
            acc = fmaf(hb, shC[d0 + i], acc);
        }
        shRed[t] = acc;
        __syncthreads();
        if (dseg == 0)
            ybuf[r * D_SSM + h * 64 + p] =
                shRed[p] + shRed[p + 64] + shRed[p + 128] + shRed[p + 192] + Dh * shX[p];
        __syncthreads();
    }
}

// ---------------- gate by silu(z), RMS-norm, convert to bf16 ----------------
__global__ void k_gate_norm(const float* __restrict__ ybuf, const float* __restrict__ zx,
                            const float* __restrict__ nw, __bf16* __restrict__ ybf) {
    const int r = blockIdx.x;
    const int t = threadIdx.x;
    __shared__ float red[256];
    float yv[8];
    float ssq = 0.f;
#pragma unroll
    for (int j = 0; j < 8; ++j) {
        int i = j * 256 + t;
        float v = ybuf[(size_t)r * D_SSM + i] * siluf(zx[(size_t)r * D_IN_PROJ + i]);
        yv[j] = v;
        ssq = fmaf(v, v, ssq);
    }
    red[t] = ssq;
    __syncthreads();
    for (int s = 128; s > 0; s >>= 1) {
        if (t < s) red[t] += red[t + s];
        __syncthreads();
    }
    const float rms = rsqrtf(red[0] * (1.0f / (float)D_SSM) + EPS);
    __syncthreads();
#pragma unroll
    for (int j = 0; j < 8; ++j) {
        int i = j * 256 + t;
        ybf[(size_t)r * D_SSM + i] = (__bf16)(yv[j] * rms * nw[i]);
    }
}

// ---------------- host-side launch ----------------
extern "C" void kernel_launch(void* const* d_in, const int* in_sizes, int n_in,
                              void* d_out, int out_size, void* d_ws, size_t ws_size,
                              hipStream_t stream) {
    const float* u       = (const float*)d_in[0];
    const float* W_in    = (const float*)d_in[1];
    const float* W_out   = (const float*)d_in[2];
    const float* dt_bias = (const float*)d_in[3];
    const float* A_log   = (const float*)d_in[4];
    const float* Dv      = (const float*)d_in[5];
    const float* B_bias  = (const float*)d_in[6];
    const float* C_bias  = (const float*)d_in[7];
    const float* Bnw     = (const float*)d_in[8];
    const float* Cnw     = (const float*)d_in[9];
    const float* nw      = (const float*)d_in[10];
    float* out = (float*)d_out;

    size_t off = 0;
    auto take = [&](size_t bytes) {
        void* p = (char*)d_ws + off;
        off += (bytes + 255) & ~(size_t)255;
        return p;
    };
    __bf16* u_bf    = (__bf16*)take((size_t)BL * D_MODEL * 2);
    __bf16* Win_bf  = (__bf16*)take((size_t)D_IN_PROJ * D_MODEL * 2);
    __bf16* Wout_bf = (__bf16*)take((size_t)D_MODEL * D_INNER * 2);
    float*  zx      = (float*) take((size_t)BL * D_IN_PROJ * 4);
    float*  sx      = (float*) take((size_t)BL * D_SSM * 4);
    float*  cum     = (float*) take((size_t)BL * 2048 * 4);
    float*  xeff    = (float*) take((size_t)BL * D_SSM * 4);
    float*  Bg      = (float*) take((size_t)BL * 128 * 4);
    float*  Cg      = (float*) take((size_t)BL * 128 * 4);
    float*  Br      = (float*) take((size_t)BL * 128 * 4);
    float*  Cr      = (float*) take((size_t)BL * 128 * 4);
    float*  Beff    = (float*) take((size_t)BL * 128 * 4);
    float*  dtb     = (float*) take((size_t)BL * 32 * 4);
    float*  dAb     = (float*) take((size_t)BL * 32 * 4);
    float*  lamb    = (float*) take((size_t)BL * 32 * 4);
    float*  lamg    = (float*) take((size_t)BL * 4);
    float*  ybuf    = (float*) take((size_t)BL * D_SSM * 4);
    __bf16* y_bf    = (__bf16*)take((size_t)BL * D_SSM * 2);

    // Stage 0: precision conversion (W matrices transposed for K-contiguous fragments)
    k_cvt <<<2048, 256, 0, stream>>>(u, u_bf, (size_t)BL * D_MODEL);
    k_cvtT<<<4096, 256, 0, stream>>>(W_in,  Win_bf,  D_MODEL, D_IN_PROJ);
    k_cvtT<<<4096, 256, 0, stream>>>(W_out, Wout_bf, D_INNER, D_MODEL);

    // Stage 1: input projection GEMM (bf16 WMMA, async-LDS double buffering)
    // grid = (1024/128) * (6464/64) = 8 * 101 = 808 (exact tiles)
    k_gemm<<<808, 256, 0, stream>>>(u_bf, Win_bf, zx, BL, D_IN_PROJ, D_MODEL);

    // Stage 2: pointwise / sequence prep
    k_prep  <<<BL, 256, 0, stream>>>(zx, B_bias, C_bias, Bnw, Cnw, dt_bias, A_log,
                                     Bg, Cg, dtb, dAb, lamb, lamg, sx);
    k_cumsum<<<16, 256, 0, stream>>>(zx, cum);
    k_rope  <<<BL, 256, 0, stream>>>(cum, Bg, Cg, Br, Cr);
    k_shift <<<BL, 256, 0, stream>>>(sx, Br, lamb, lamg, xeff, Beff);

    // Stage 3: sequential SSM scan, one block per (batch, head)
    k_scan<<<B_SZ * NHEADS, 256, 0, stream>>>(dAb, dtb, xeff, Beff, Cr, Dv, ybuf);

    // Stage 4: gate + RMS norm -> bf16
    k_gate_norm<<<BL, 256, 0, stream>>>(ybuf, zx, nw, y_bf);

    // Stage 5: output projection GEMM -> d_out (f32)
    // grid = (1024/128) * (1024/64) = 8 * 16 = 128 (exact tiles)
    k_gemm<<<128, 256, 0, stream>>>(y_bf, Wout_bf, out, BL, D_MODEL, D_INNER);
}
